// MoEBiEncoder_6511170421321
// MI455X (gfx1250) — matrix operand order repaired and checked by reading, not compiled, not run on hardware
//
#include <hip/hip_runtime.h>
#include <stdint.h>

#define BATCH 16384
#define NEXP  16
#define DIM   768
#define HID   384
#define LDA   776   // 768+8 bf16, 16B-aligned rows, conflict-free
#define LDH   392   // 384+8

typedef __attribute__((ext_vector_type(16))) __bf16 v16bf;
typedef __attribute__((ext_vector_type(8)))  __bf16 v8bf;
typedef __attribute__((ext_vector_type(8)))  float  v8f;

// ---------------- routing ----------------

__global__ void init_counts(int* __restrict__ counts) {
  if (threadIdx.x < NEXP) counts[threadIdx.x] = 0;
}

__global__ void route_kernel(const float* __restrict__ logits,
                             float* __restrict__ gate,
                             int* __restrict__ eid,
                             int* __restrict__ counts) {
  int b = blockIdx.x * blockDim.x + threadIdx.x;
  if (b >= BATCH) return;
  const float* lp = logits + (size_t)b * NEXP;
  float l[NEXP];
  float mx = -3.402823466e38f; int am = 0;
#pragma unroll
  for (int i = 0; i < NEXP; ++i) { l[i] = lp[i]; if (l[i] > mx) { mx = l[i]; am = i; } }
  float s = 0.f;
#pragma unroll
  for (int i = 0; i < NEXP; ++i) s += __expf((l[i] - mx) * 0.1f);
  gate[b] = 1.0f / s;          // softmax value at argmax
  eid[b]  = am;
  atomicAdd(&counts[am], 1);
}

__global__ void prefix_kernel(const int* __restrict__ counts, int* __restrict__ cursors) {
  if (threadIdx.x == 0) {
    int run = 0;
    for (int c = 0; c < NEXP; ++c) { cursors[c] = run; run += counts[c]; }
  }
}

__global__ void scatter_kernel(const int* __restrict__ eid,
                               int* __restrict__ cursors,
                               int* __restrict__ rowidx) {
  int b = blockIdx.x * blockDim.x + threadIdx.x;
  if (b >= BATCH) return;
  int pos = atomicAdd(&cursors[eid[b]], 1);
  rowidx[pos] = b;
}

// ---------- weight fp32 -> bf16 transpose (K made contiguous for B-fragments) ----------
// src: [NEXP][R][C] fp32 (C contiguous) -> dst: [NEXP][C][R] bf16 (R contiguous)
__global__ void transpose_convert(const float* __restrict__ src, __bf16* __restrict__ dst,
                                  int R, int C) {
  __shared__ float tile[32][33];
  int z = blockIdx.z;
  const float* s = src + (size_t)z * R * C;
  __bf16*      d = dst + (size_t)z * R * C;
  int c0 = blockIdx.x * 32, r0 = blockIdx.y * 32;
  int x = threadIdx.x;
  for (int j = threadIdx.y; j < 32; j += 8)
    tile[j][x] = s[(size_t)(r0 + j) * C + c0 + x];
  __syncthreads();
  for (int j = threadIdx.y; j < 32; j += 8)
    d[(size_t)(c0 + j) * R + r0 + x] = (__bf16)tile[x][j];
}

// ---------------- grouped-GEMM expert FFN ----------------

__launch_bounds__(128, 1)
__global__ void ffn_kernel(const float* __restrict__ emb,
                           const float* __restrict__ b1,
                           const float* __restrict__ b4,
                           const float* __restrict__ gate,
                           const int*   __restrict__ counts,
                           const int*   __restrict__ rowidx,
                           const __bf16* __restrict__ w1t,   // [NEXP][HID][DIM]
                           const __bf16* __restrict__ w4t,   // [NEXP][DIM][HID]
                           float* __restrict__ out) {
  // ---- map block -> (expert, local tile), recompute offsets from counts ----
  int cnt[NEXP];
#pragma unroll
  for (int c = 0; c < NEXP; ++c) cnt[c] = counts[c];
  int bid = blockIdx.x;
  int e = -1, lt = 0, base = 0, tb = 0;
#pragma unroll
  for (int c = 0; c < NEXP; ++c) {
    int pt = (cnt[c] + 15) >> 4;
    if (e < 0) {
      if (bid < tb + pt) { e = c; lt = bid - tb; }
      else base += cnt[c];
    }
    tb += pt;
  }
  if (e < 0) return;
  int tileStart = base + lt * 16;

  const int tid    = threadIdx.x;
  const int wave   = tid >> 5;
  const int lane   = tid & 31;
  const int laneLo = lane & 15;
  const int laneHi = lane >> 4;

  __shared__ __bf16 Alds[16 * LDA];   // 16 x 768 embedding tile (bf16)
  __shared__ __bf16 Hlds[16 * LDH];   // 16 x 384 hidden tile (bf16)
  __shared__ float  s_red[16][4];
  __shared__ float  s_inv1[16];
  __shared__ float  s_inv2[16];
  __shared__ float  s_gate[16];
  __shared__ int    s_row[16];
  __shared__ int    s_valid[16];

  if (tid < 16) {
    int m  = tid;
    int ok = (lt * 16 + m) < cnt[e];
    int r  = ok ? rowidx[tileStart + m] : 0;
    s_row[m]   = r;
    s_valid[m] = ok;
    s_gate[m]  = ok ? gate[r] : 0.f;
  }
  __syncthreads();

  // ---- stage gathered embedding rows -> bf16 LDS (8 threads/row, coalesced float4) ----
  {
    int m  = tid >> 3;
    int cg = tid & 7;
    const float* src = emb + (size_t)s_row[m] * DIM;
    bool ok = s_valid[m] != 0;
#pragma unroll
    for (int k = 0; k < 24; ++k) {
      int col = cg * 4 + k * 32;
      float4 v = ok ? *(const float4*)(src + col) : make_float4(0.f, 0.f, 0.f, 0.f);
      __bf16* dst = &Alds[m * LDA + col];
      dst[0] = (__bf16)v.x; dst[1] = (__bf16)v.y;
      dst[2] = (__bf16)v.z; dst[3] = (__bf16)v.w;
    }
  }
  __syncthreads();

  const v8f zero8 = {0.f, 0.f, 0.f, 0.f, 0.f, 0.f, 0.f, 0.f};

  // ================= GEMM1: h = relu(X @ W1 + b1), 16x384, K=768 =================
  const __bf16* w1e = w1t + (size_t)e * HID * DIM;
  v8f acc1[6];
#pragma unroll
  for (int t = 0; t < 6; ++t) acc1[t] = zero8;

  for (int k0 = 0; k0 < DIM; k0 += 32) {
    int ka = k0 + (laneHi ? 16 : 0);
    v8bf a0 = *(const v8bf*)&Alds[laneLo * LDA + ka];
    v8bf a1 = *(const v8bf*)&Alds[laneLo * LDA + ka + 8];
    v16bf afrag = __builtin_shufflevector(a0, a1, 0,1,2,3,4,5,6,7,8,9,10,11,12,13,14,15);
    int kb = k0 + (laneHi ? 16 : 0);
#pragma unroll
    for (int t = 0; t < 6; ++t) {
      int n = wave * 96 + t * 16 + laneLo;
      v16bf bfrag = *(const v16bf*)(w1e + (size_t)n * DIM + kb);
      acc1[t] = __builtin_amdgcn_wmma_f32_16x16x32_bf16(
          false, afrag, false, bfrag, (short)0, acc1[t], false, false);
    }
  }
#pragma unroll
  for (int t = 0; t < 6; ++t) {
    int n = wave * 96 + t * 16 + laneLo;
    float bias = b1[e * HID + n];
#pragma unroll
    for (int r = 0; r < 8; ++r) {
      int m = r + 8 * laneHi;
      float v = acc1[t][r] + bias;
      Hlds[m * LDH + n] = (__bf16)fmaxf(v, 0.f);
    }
  }
  __syncthreads();

  // ================= GEMM2: out = h @ W4 + b4, 16x768, K=384 =================
  const __bf16* w4e = w4t + (size_t)e * DIM * HID;
  v8f acc2[12];
#pragma unroll
  for (int t = 0; t < 12; ++t) acc2[t] = zero8;

  for (int k0 = 0; k0 < HID; k0 += 32) {
    int ka = k0 + (laneHi ? 16 : 0);
    v8bf a0 = *(const v8bf*)&Hlds[laneLo * LDH + ka];
    v8bf a1 = *(const v8bf*)&Hlds[laneLo * LDH + ka + 8];
    v16bf afrag = __builtin_shufflevector(a0, a1, 0,1,2,3,4,5,6,7,8,9,10,11,12,13,14,15);
    int kb = k0 + (laneHi ? 16 : 0);
#pragma unroll
    for (int t = 0; t < 12; ++t) {
      int n = wave * 192 + t * 16 + laneLo;
      v16bf bfrag = *(const v16bf*)(w4e + (size_t)n * HID + kb);
      acc2[t] = __builtin_amdgcn_wmma_f32_16x16x32_bf16(
          false, afrag, false, bfrag, (short)0, acc2[t], false, false);
    }
  }

  // ---- epilogue: +b4, *gate ----
  float gv[8];
#pragma unroll
  for (int r = 0; r < 8; ++r) gv[r] = s_gate[r + 8 * laneHi];
#pragma unroll
  for (int t = 0; t < 12; ++t) {
    int n = wave * 192 + t * 16 + laneLo;
    float b4v = b4[e * DIM + n];
#pragma unroll
    for (int r = 0; r < 8; ++r) acc2[t][r] = gv[r] * (acc2[t][r] + b4v);
  }

  // ---- norm1: per-row ||comb|| ----
#pragma unroll
  for (int r = 0; r < 8; ++r) {
    float v = 0.f;
#pragma unroll
    for (int t = 0; t < 12; ++t) { float x = acc2[t][r]; v += x * x; }
    v += __shfl_xor(v, 1); v += __shfl_xor(v, 2);
    v += __shfl_xor(v, 4); v += __shfl_xor(v, 8);
    if (laneLo == 0) s_red[r + 8 * laneHi][wave] = v;
  }
  __syncthreads();
  if (tid < 16) {
    float s = s_red[tid][0] + s_red[tid][1] + s_red[tid][2] + s_red[tid][3];
    s_inv1[tid] = 1.0f / fmaxf(sqrtf(s), 1e-6f);
  }
  __syncthreads();

  // ---- norm2: || comb/n1 + emb || (emb reloaded fp32, L2-hot) ----
  float i1[8];
  const float* ep[8];
#pragma unroll
  for (int r = 0; r < 8; ++r) {
    int m = r + 8 * laneHi;
    i1[r] = s_inv1[m];
    ep[r] = emb + (size_t)s_row[m] * DIM;
  }
#pragma unroll
  for (int r = 0; r < 8; ++r) {
    float v = 0.f;
#pragma unroll
    for (int t = 0; t < 12; ++t) {
      int n = wave * 192 + t * 16 + laneLo;
      float u = acc2[t][r] * i1[r] + ep[r][n];
      v += u * u;
    }
    v += __shfl_xor(v, 1); v += __shfl_xor(v, 2);
    v += __shfl_xor(v, 4); v += __shfl_xor(v, 8);
    if (laneLo == 0) s_red[r + 8 * laneHi][wave] = v;
  }
  __syncthreads();
  if (tid < 16) {
    float s = s_red[tid][0] + s_red[tid][1] + s_red[tid][2] + s_red[tid][3];
    s_inv2[tid] = 1.0f / fmaxf(sqrtf(s), 1e-12f);
  }
  __syncthreads();

  // ---- final scatter store ----
  float i2[8]; int vl[8]; float* op[8];
#pragma unroll
  for (int r = 0; r < 8; ++r) {
    int m = r + 8 * laneHi;
    i2[r] = s_inv2[m];
    vl[r] = s_valid[m];
    op[r] = out + (size_t)s_row[m] * DIM;
  }
#pragma unroll
  for (int t = 0; t < 12; ++t) {
    int n = wave * 192 + t * 16 + laneLo;
#pragma unroll
    for (int r = 0; r < 8; ++r) {
      if (vl[r]) {
        float u = acc2[t][r] * i1[r] + ep[r][n];
        op[r][n] = u * i2[r];
      }
    }
  }
}

// ---------------- launch ----------------

extern "C" void kernel_launch(void* const* d_in, const int* in_sizes, int n_in,
                              void* d_out, int out_size, void* d_ws, size_t ws_size,
                              hipStream_t stream) {
  const float* emb    = (const float*)d_in[0];
  const float* logits = (const float*)d_in[1];
  const float* w1     = (const float*)d_in[2];
  const float* b1     = (const float*)d_in[3];
  const float* w4     = (const float*)d_in[4];
  const float* b4     = (const float*)d_in[5];
  float* out = (float*)d_out;

  uint8_t* w = (uint8_t*)d_ws;
  int*   counts  = (int*)w;
  int*   cursors = counts + 16;
  int*   eid     = cursors + 16;
  float* gate    = (float*)(eid + BATCH);
  int*   rowidx  = (int*)(gate + BATCH);
  uintptr_t p = (uintptr_t)(rowidx + BATCH);
  p = (p + 255) & ~(uintptr_t)255;
  __bf16* w1t = (__bf16*)p;
  __bf16* w4t = w1t + (size_t)NEXP * HID * DIM;

  init_counts<<<1, 64, 0, stream>>>(counts);
  route_kernel<<<BATCH / 256, 256, 0, stream>>>(logits, gate, eid, counts);
  prefix_kernel<<<1, 32, 0, stream>>>(counts, cursors);
  scatter_kernel<<<BATCH / 256, 256, 0, stream>>>(eid, cursors, rowidx);
  // w1: [16][768][384] -> w1t [16][384][768]
  transpose_convert<<<dim3(HID / 32, DIM / 32, NEXP), dim3(32, 8), 0, stream>>>(w1, w1t, DIM, HID);
  // w4: [16][384][768] -> w4t [16][768][384]
  transpose_convert<<<dim3(DIM / 32, HID / 32, NEXP), dim3(32, 8), 0, stream>>>(w4, w4t, HID, DIM);
  ffn_kernel<<<BATCH / 16 + NEXP, 128, 0, stream>>>(emb, b1, b4, gate, counts, rowidx,
                                                    w1t, w4t, out);
}
